// ViscoelasticMaterialModel_1486058684504
// MI455X (gfx1250) — matrix operand order repaired and checked by reading, not compiled, not run on hardware
//
#include <hip/hip_runtime.h>
#include <hip/hip_bf16.h>

// ---------------------------------------------------------------------------
// Viscoelastic material model (FNF + internal-variable scan) for gfx1250.
// Kernel 1 (fnf_kernel): 4 Fourier-neural-functionals.
//   - truncated rFFT  G = basis@h   ([16pad,512]@[512,32]) via chained WMMA,
//     B-operand fed from a transposed f16 mirror (contiguous ds_load_b128)
//   - pointwise       P = h@pw_w    ([512,32]@[32,32])      via WMMA
//   - irFFT expansion + bias + GELU fused into WMMA result lanes; both h
//     layouts updated in place (hhT via one packed ds_store_b128 per tile)
// Kernel 2 (visco_scan): sequential T=100 scan, one wave32 per batch chain.
// ---------------------------------------------------------------------------

typedef _Float16 v16h __attribute__((ext_vector_type(16)));
typedef _Float16 v8h  __attribute__((ext_vector_type(8)));
typedef float    v8f  __attribute__((ext_vector_type(8)));

#define WIDTH   32
#define NSEQ    512
#define MODES   4
#define NLAYERS 3
#define WFIN    64
#define TWO_PI_OVER_N 0.012271846303085130f // 2*pi/512

__device__ __forceinline__ float gelu_f(float x) {
  const float k0 = 0.7978845608028654f;   // sqrt(2/pi)
  const float k1 = 0.044715f;
  return 0.5f * x * (1.0f + tanhf(k0 * (x + k1 * x * x * x)));
}

// ---------------------------------------------------------------------------
// FNF kernel: grid (B, 4), block 256 (8 waves). blockIdx.y selects which FNF.
// Param buffer layout = JAX pytree (sorted dict keys):
//   fc0_b[32], fc0_w[c,32], fc1_b[64], fc1_w[32,64], fc2_b[1], fc2_w[64,1],
//   pw_b[3,32], pw_w[3,32,32], spec_i[3,32,32,4], spec_r[3,32,32,4]
// ---------------------------------------------------------------------------
__global__ __launch_bounds__(256) void fnf_kernel(
    const float* __restrict__ E, const float* __restrict__ NU,
    const float* __restrict__ P0, const float* __restrict__ P1,
    const float* __restrict__ P2, const float* __restrict__ P3,
    float* __restrict__ fvals, int B)
{
  const int b  = blockIdx.x;
  const int fi = blockIdx.y;
  const float* P = (fi == 0) ? P0 : (fi == 1) ? P1 : (fi == 2) ? P2 : P3;
  const int c = (fi == 2) ? 1 : 2;   // p_d1 sees only the nu channel

  const int OFF_FC0B  = 0;
  const int OFF_FC0W  = 32;
  const int OFF_FC1B  = 32 + 32 * c;
  const int OFF_FC1W  = OFF_FC1B + 64;
  const int OFF_FC2B  = OFF_FC1W + 2048;
  const int OFF_FC2W  = OFF_FC2B + 1;
  const int OFF_PWB   = OFF_FC2W + 64;
  const int OFF_PWW   = OFF_PWB + 96;
  const int OFF_SPECI = OFF_PWW + 3072;
  const int OFF_SPECR = OFF_SPECI + 12288;

  // LDS (static): hh 32K + hhT 32K + bas16 16K + pwhT 2K + GPart 8K + misc
  //               = 95,744 B  (CDNA5 allows up to 320KB per workgroup)
  __shared__ __align__(16) _Float16 hh [NSEQ * WIDTH];    // h, row-major [n][w]
  __shared__ __align__(16) _Float16 hhT[WIDTH * NSEQ];    // h, transposed [w][n]
  __shared__ __align__(16) _Float16 bas16[16 * NSEQ];     // DFT basis rows:
                                                          // 0-3 cos(m), 4-7 sin(m), 8-15 zero
  __shared__ __align__(16) _Float16 pwhT[WIDTH * WIDTH];  // pw_w^T staged f16
  __shared__ float pwb_s[WIDTH];
  __shared__ float Gc[MODES * WIDTH], Gs[MODES * WIDTH];  // rfft modes
  __shared__ float Or_[MODES * WIDTH], Oi_[MODES * WIDTH];
  __shared__ float GPart[4 * 16 * WIDTH];                 // K-split WMMA partials
  __shared__ float psum[8 * WIDTH];
  __shared__ float gvec[WIDTH];
  __shared__ float t1s[WFIN];

  const int tid = threadIdx.x;

  // --- f16 DFT basis table (A-operand for DFT-WMMA + irfft expansion) ---
  for (int i = tid; i < 16 * NSEQ; i += 256) {
    int row = i >> 9, n = i & 511;
    float v = 0.f;
    if (row < 8) {
      int m = row & 3;
      float s, co;
      __sincosf(TWO_PI_OVER_N * (float)(m * n), &s, &co);
      v = (row < 4) ? co : s;
    }
    bas16[i] = (_Float16)v;
  }

  // --- lifting: h[n,w] = sum_c x[c,n]*fc0_w[c,w] + fc0_b[w] (both layouts) ---
  const float* Eb = E  + (size_t)b * NSEQ;
  const float* Nb = NU + (size_t)b * NSEQ;
  for (int i = tid; i < NSEQ * WIDTH; i += 256) {
    int n = i >> 5, w = i & 31;
    float acc = P[OFF_FC0B + w];
    if (c == 2) acc += Eb[n] * P[OFF_FC0W + w] + Nb[n] * P[OFF_FC0W + 32 + w];
    else        acc += Nb[n] * P[OFF_FC0W + w];
    _Float16 h16 = (_Float16)acc;
    hh[i] = h16;
    hhT[(w << 9) + n] = h16;
  }

  const int wv   = tid >> 5, lane = tid & 31;
  const int g    = lane >> 4, l15 = lane & 15;

  for (int l = 0; l < NLAYERS; ++l) {
    __syncthreads();
    // --- truncated rfft via WMMA: G[16pad,32] = bas16[16,512] @ h[512,32] ---
    // wave wv: N-tile (wv&1), K-quarter (wv>>1): 4 chained K-step WMMAs.
    {
      int nt = wv & 1, kg = wv >> 1;
      int wcol = (nt << 4) + l15;
      v8f cacc = {};
#pragma unroll
      for (int kk = 0; kk < 4; ++kk) {
        int k = (kg << 2) + kk;               // 32-wide K chunk
        // A: basis rows 0..15, cols k*32..k*32+31 (ISA 16-bit A layout)
        const _Float16* ap = &bas16[(l15 << 9) + (k << 5) + (g << 3)];
        v8h a0 = *(const v8h*)ap;
        v8h a1 = *(const v8h*)(ap + 16);
        v16h aa;
#pragma unroll
        for (int j = 0; j < 8; ++j) { aa[j] = a0[j]; aa[j + 8] = a1[j]; }
        // B: h rows k*32+g*8+j and k*32+16+g*8+j (K), column wcol -> hhT
        const _Float16* bp = &hhT[(wcol << 9) + (k << 5) + (g << 3)];
        v8h b0 = *(const v8h*)bp;
        v8h b1 = *(const v8h*)(bp + 16);
        v16h bb;
#pragma unroll
        for (int j = 0; j < 8; ++j) { bb[j] = b0[j]; bb[j + 8] = b1[j]; }
        cacc = __builtin_amdgcn_wmma_f32_16x16x32_f16(
            false, aa, false, bb, (short)0, cacc, false, false);
      }
#pragma unroll
      for (int r = 0; r < 8; ++r)
        GPart[(kg << 9) + (((g << 3) + r) << 5) + wcol] = cacc[r];
    }
    __syncthreads();
    // --- reduce 4 K-partials -> Gc (cos rows 0-3) / Gs (sin rows 4-7) ---
    for (int idx = tid; idx < 512; idx += 256) {
      float s = GPart[idx] + GPart[512 + idx] + GPart[1024 + idx] + GPart[1536 + idx];
      int row = idx >> 5, wc2 = idx & 31;
      if (row < 4)      Gc[(row << 5) + wc2] = s;
      else if (row < 8) Gs[((row - 4) << 5) + wc2] = s;
    }
    __syncthreads();
    // --- complex spectral mix (X = Gc - i*Gs ; O = X*(Wr + i*Wi)) + staging ---
    {
      int seg = tid >> 7, m = (tid >> 5) & 3, o = tid & 31;
      const float* Wr = P + OFF_SPECR + l * 4096;   // [i,o,m] strides 128,4,1
      const float* Wi = P + OFF_SPECI + l * 4096;
      float acc = 0.f;
      for (int i2 = 0; i2 < WIDTH; ++i2) {
        float wr = Wr[i2 * 128 + o * 4 + m];
        float wi = Wi[i2 * 128 + o * 4 + m];
        float gc = Gc[(m << 5) + i2], gs = Gs[(m << 5) + i2];
        acc += (seg == 0) ? (gc * wr + gs * wi) : (gc * wi - gs * wr);
      }
      if (seg == 0) Or_[(m << 5) + o] = acc;
      else          Oi_[(m << 5) + o] = acc;
      // stage pw_w[l]^T into LDS f16: pwhT[o,i]
      const float* PW = P + OFF_PWW + l * 1024;
      for (int e2 = 0; e2 < 4; ++e2) {
        int idx = (tid << 2) + e2;
        int oo = idx >> 5, ii = idx & 31;
        pwhT[(oo << 5) + ii] = (_Float16)PW[(ii << 5) + oo];
      }
      if (tid < 32) pwb_s[tid] = P[OFF_PWB + l * 32 + tid];
    }
    __syncthreads();
    // --- pointwise matmul via WMMA + irfft expansion + GELU ---
    {
      v16h bfrag[2];
#pragma unroll
      for (int nt = 0; nt < 2; ++nt) {
        const _Float16* rp = &pwhT[(((nt << 4) + l15) << 5) + (g << 3)];
        v8h b0 = *(const v8h*)rp;
        v8h b1 = *(const v8h*)(rp + 16);
        v16h bb;
#pragma unroll
        for (int j = 0; j < 8; ++j) { bb[j] = b0[j]; bb[j + 8] = b1[j]; }
        bfrag[nt] = bb;
      }
#pragma unroll
      for (int mt4 = 0; mt4 < 4; ++mt4) {
        int mt = (wv << 2) + mt4;             // wave owns rows [wv*64, wv*64+64)
        const _Float16* rp = &hh[(((mt << 4) + l15) << 5) + (g << 3)];
        v8h a0 = *(const v8h*)rp;
        v8h a1 = *(const v8h*)(rp + 16);
        v16h aa;
#pragma unroll
        for (int j = 0; j < 8; ++j) { aa[j] = a0[j]; aa[j + 8] = a1[j]; }
        v8f cf[2];
#pragma unroll
        for (int nt = 0; nt < 2; ++nt) {
          v8f cz = {};
          cf[nt] = __builtin_amdgcn_wmma_f32_16x16x32_f16(
              false, aa, false, bfrag[nt], (short)0, cz, false, false);
        }
#pragma unroll
        for (int nt = 0; nt < 2; ++nt) {
          int wcol = (nt << 4) + l15;
          float pb  = pwb_s[wcol];
          float o0  = Or_[wcol];                    // DC (imag dropped by irfft)
          float o1r = Or_[32 + wcol], o1i = Oi_[32 + wcol];
          float o2r = Or_[64 + wcol], o2i = Oi_[64 + wcol];
          float o3r = Or_[96 + wcol], o3i = Oi_[96 + wcol];
          v8h pack;
#pragma unroll
          for (int r = 0; r < 8; ++r) {
            int npos = (mt << 4) + (g << 3) + r;    // C/D layout: M = r + 8*g
            float spec = o0
              + 2.f * (o1r * (float)bas16[ 512 + npos] - o1i * (float)bas16[2560 + npos])
              + 2.f * (o2r * (float)bas16[1024 + npos] - o2i * (float)bas16[3072 + npos])
              + 2.f * (o3r * (float)bas16[1536 + npos] - o3i * (float)bas16[3584 + npos]);
            float val = cf[nt][r] + pb + spec * (1.0f / 512.0f);
            if (l < NLAYERS - 1) val = gelu_f(val);
            _Float16 h16 = (_Float16)val;
            hh[(npos << 5) + wcol] = h16;           // wave-private rows: no race
            pack[r] = h16;
          }
          // transposed mirror: npos run is contiguous -> one ds_store_b128
          *(v8h*)&hhT[(wcol << 9) + (mt << 4) + (g << 3)] = pack;
        }
      }
    }
  }
  __syncthreads();
  // --- functional pooling (contiguous scans of hhT) + MLP head ---
  {
    int w = tid & 31, ch = tid >> 5;
    const _Float16* rp = &hhT[(w << 9) + ch * 64];
    float acc = 0.f;
    for (int n = 0; n < 64; ++n) acc += (float)rp[n];
    psum[(ch << 5) + w] = acc;
  }
  __syncthreads();
  if (tid < 32) {
    float acc = 0.f;
    for (int ch = 0; ch < 8; ++ch) acc += psum[(ch << 5) + tid];
    gvec[tid] = acc * (1.0f / 512.0f);
  }
  __syncthreads();
  if (tid < WFIN) {
    float acc = P[OFF_FC1B + tid];
    for (int w = 0; w < WIDTH; ++w) acc += gvec[w] * P[OFF_FC1W + w * WFIN + tid];
    t1s[tid] = gelu_f(acc);
  }
  __syncthreads();
  if (tid == 0) {
    float acc = P[OFF_FC2B];
    for (int j = 0; j < WFIN; ++j) acc += t1s[j] * P[OFF_FC2W + j];
    fvals[(size_t)fi * B + b] = acc;
  }
}

// ---------------------------------------------------------------------------
// Scan kernel: one wave32 per batch chain; lane-parallel hidden units with
// shfl_xor butterflies. Energy net input is (u, xi, f1, f2) — xi feedback
// makes each chain strictly sequential over T.
// Param layouts (sorted keys): enn: b1[100],b2[1],w1[4,100],w2[100]
//                              dnn: b1[50],b2[1],w1[50],w2[50]
//                              cvx: b0[50],b1[1],w0[50],w1[50]
// ---------------------------------------------------------------------------
__global__ __launch_bounds__(256) void visco_scan(
    const float* __restrict__ e, const float* __restrict__ edot,
    const float* __restrict__ enn, const float* __restrict__ dnn,
    const float* __restrict__ cvx, const float* __restrict__ fvals,
    float* __restrict__ out, int B, int T)
{
  __shared__ float sp[601 + 151 + 151];
  for (int i = threadIdx.x; i < 903; i += blockDim.x)
    sp[i] = (i < 601) ? enn[i] : (i < 752 ? dnn[i - 601] : cvx[i - 752]);
  __syncthreads();

  const float* eb1 = sp;           // enn b1 [100]
  const float* ew1 = sp + 101;     // enn w1 [4,100]
  const float* ew2 = sp + 501;     // enn w2 [100]
  const float* db1 = sp + 601;     // dnn b1 [50]
  const float* dw1 = sp + 652;     // dnn w1 [50]
  const float* dw2 = sp + 702;     // dnn w2 [50]
  const float* cb0 = sp + 752;     // cvx b0 [50]
  const float* cw0 = sp + 803;     // cvx w0 [50]
  const float* cw1 = sp + 853;     // cvx w1 [50]

  int wave = (blockIdx.x * blockDim.x + threadIdx.x) >> 5;
  int lane = threadIdx.x & 31;
  if (wave >= B) return;
  int b = wave;

  float f1 = fvals[b],         f2 = fvals[B + b];
  float g1 = fvals[2 * B + b], g2 = fvals[3 * B + b];
  float xi = 0.f;

  for (int t = 0; t < T; ++t) {
    float u = e[(size_t)b * T + t];
    float v = edot[(size_t)b * T + t];

    // d(energy)/du, d(energy)/dxi : mlp(p_enn, [u, xi, f1, f2])
    float seq = 0.f, dd = 0.f;
    for (int h = lane; h < 100; h += 32) {
      float z = u * ew1[h] + xi * ew1[100 + h] + f1 * ew1[200 + h]
              + f2 * ew1[300 + h] + eb1[h];
      if (z > 0.f) {
        float a = 2.f * z * ew2[h];      // d(sq_relu)/dz * w2
        seq += a * ew1[h];
        dd  += a * ew1[100 + h];
      }
    }
#pragma unroll
    for (int off = 16; off > 0; off >>= 1) {
      seq += __shfl_xor(seq, off, 32);
      dd  += __shfl_xor(dd, off, 32);
    }

    // s_neq = -0.5*g1 * d mlp_dnn(pdot)/dpdot, pdot = e_dot
    float sn = 0.f;
    for (int h = lane; h < 50; h += 32) {
      float z = v * dw1[h] + db1[h];
      if (z > 0.f) sn += 2.f * z * dw1[h] * dw2[h];
    }
#pragma unroll
    for (int off = 16; off > 0; off >>= 1) sn += __shfl_xor(sn, off, 32);
    float s_neq = -0.5f * g1 * sn;

    // kin = 0.5*g2 * d convex(q)/dq at q = -dd  (w1 squared -> convexity)
    float q = -dd, kv = 0.f;
    for (int h = lane; h < 50; h += 32) {
      float z = q * cw0[h] + cb0[h];
      if (z > 0.f) kv += 2.f * z * cw0[h] * cw1[h] * cw1[h];
    }
#pragma unroll
    for (int off = 16; off > 0; off >>= 1) kv += __shfl_xor(kv, off, 32);
    float kin = 0.5f * g2 * kv;

    if (lane == 0) {
      out[(size_t)b * T + t] = seq - s_neq;               // stress
      out[(size_t)B * T + (size_t)b * T + t] = xi;        // xi (pre-update)
    }
    xi += 0.01f * kin;   // DT
  }
}

// ---------------------------------------------------------------------------
extern "C" void kernel_launch(void* const* d_in, const int* in_sizes, int n_in,
                              void* d_out, int out_size, void* d_ws, size_t ws_size,
                              hipStream_t stream) {
  const float* e    = (const float*)d_in[0];
  const float* edot = (const float*)d_in[1];
  const float* E    = (const float*)d_in[2];
  const float* NU   = (const float*)d_in[3];
  const float* p_e1 = (const float*)d_in[4];
  const float* p_e2 = (const float*)d_in[5];
  const float* p_d1 = (const float*)d_in[6];
  const float* p_d2 = (const float*)d_in[7];
  const float* enn  = (const float*)d_in[8];
  const float* dnn  = (const float*)d_in[9];
  const float* cvx  = (const float*)d_in[10];

  int B = in_sizes[2] / NSEQ;        // E is [B, 512]
  int T = in_sizes[0] / B;           // e is [B, T, 1]
  float* fvals = (float*)d_ws;       // [4, B] scratch

  dim3 gridA(B, 4, 1);
  fnf_kernel<<<gridA, 256, 0, stream>>>(E, NU, p_e1, p_e2, p_d1, p_d2, fvals, B);

  int nblk = (B + 7) / 8;            // 8 waves (chains) per 256-thread block
  visco_scan<<<nblk, 256, 0, stream>>>(e, edot, enn, dnn, cvx, fvals,
                                       (float*)d_out, B, T);
}
